// MultiHeadAttention_82085414961899
// MI455X (gfx1250) — matrix operand order repaired and checked by reading, not compiled, not run on hardware
//
#include <hip/hip_runtime.h>
#include <hip/hip_bf16.h>

// ---------------------------------------------------------------------------
// MHA for MI455X (gfx1250): bf16 WMMA everywhere, flash-attention inner loop,
// LDS-staged weight panels for the two GEMMs.
//   DIM=768, NHEAD=12, HD=64, B=8, SEQ=1024, SCALE=1/8
// ---------------------------------------------------------------------------

typedef __attribute__((ext_vector_type(16))) __bf16 v16bf;
typedef __attribute__((ext_vector_type(8)))  float  v8f;

#define DIM    768
#define NHEAD  12
#define HD     64
#define BATCH  8
#define SEQ    1024
#define MTOK   (BATCH*SEQ)          // 8192
#define QKVO   (3*DIM)              // 2304
#define KC     32                   // K chunk per pipeline stage
#define LDB    40                   // padded LDS row stride (elems) -> conflict-free b128

__device__ __forceinline__ unsigned short f2bf(float f) {
  unsigned int u = __float_as_uint(f);
  u += 0x7FFFu + ((u >> 16) & 1u);          // round-to-nearest-even
  return (unsigned short)(u >> 16);
}

union Frag {                 // one WMMA 16-bit operand (16 bf16 = 8 VGPRs)
  v16bf v;
  uint4 u[2];
  unsigned short s[16];
};
union Pack8 { unsigned short s[8]; uint4 u; };

__device__ __forceinline__ v8f wmma_bf16(const Frag& a, const Frag& b, v8f c) {
  return __builtin_amdgcn_wmma_f32_16x16x32_bf16(false, a.v, false, b.v,
                                                 (short)0, c, false, false);
}

// ---------------------------------------------------------------------------
// fp32 -> bf16 convert, 4-wide
// ---------------------------------------------------------------------------
__global__ __launch_bounds__(256)
void cvt_bf16_kernel(const float4* __restrict__ in, ushort4* __restrict__ out, int n4) {
  int i = blockIdx.x * 256 + threadIdx.x;
  if (i < n4) {
    float4 f = in[i];
    ushort4 o;
    o.x = f2bf(f.x); o.y = f2bf(f.y); o.z = f2bf(f.z); o.w = f2bf(f.w);
    out[i] = o;
  }
}

// ---------------------------------------------------------------------------
// Shared GEMM core: block computes a 256(M) x 64(N) tile of  A(bf16) * W^T.
//   - W chunk (64 x 32) staged in LDS once per block, double buffered
//   - each wave: 32(M) x 64(N), 8 WMMAs per k-step from 2 A + 4 LDS B frags
// Accumulators returned in acc[sub][s]; caller handles the epilogue.
// ---------------------------------------------------------------------------
__device__ __forceinline__ void gemm_core(const unsigned short* __restrict__ ab,
                                          const unsigned short* __restrict__ wb,
                                          int row0, int obase,
                                          unsigned short* Bs,     // [2][64*LDB]
                                          v8f acc[2][4]) {
  const int lane = threadIdx.x & 31;
  const int h = lane >> 4, ln = lane & 15;

  // cooperative W-chunk loader: thread t -> row t/4, kpart (t%4)*8 (16B)
  const int brow = threadIdx.x >> 2;
  const int bk   = (threadIdx.x & 3) * 8;
  const unsigned short* bsrc = wb + (size_t)(obase + brow) * DIM + bk;
  unsigned short* bdst = Bs + brow * LDB + bk;          // +64*LDB for buffer 1

  const unsigned short* arow0 = ab + (size_t)(row0 + ln) * DIM;
  const unsigned short* arow1 = ab + (size_t)(row0 + 16 + ln) * DIM;

  // stage chunk 0
  *(uint4*)(bdst) = *(const uint4*)(bsrc);
  __syncthreads();

  for (int k0 = 0; k0 < DIM; k0 += KC) {
    const int cur = (k0 >> 5) & 1;
    const bool more = (k0 + KC) < DIM;
    uint4 bnext;
    if (more) bnext = *(const uint4*)(bsrc + k0 + KC);  // global prefetch

    Frag a0, a1;
    a0.u[0] = *(const uint4*)(arow0 + k0 + 8 * h);
    a0.u[1] = *(const uint4*)(arow0 + k0 + 8 * h + 16);
    a1.u[0] = *(const uint4*)(arow1 + k0 + 8 * h);
    a1.u[1] = *(const uint4*)(arow1 + k0 + 8 * h + 16);

    const unsigned short* bbuf = Bs + cur * (64 * LDB);
#pragma unroll
    for (int s = 0; s < 4; ++s) {
      Frag b;                                           // B: 32x16, K 8h..8h+7 and 16+8h..16+8h+7 by half
      const unsigned short* bp = bbuf + (s * 16 + ln) * LDB + 8 * h;
      b.u[0] = *(const uint4*)(bp);
      b.u[1] = *(const uint4*)(bp + 16);
      acc[0][s] = wmma_bf16(a0, b, acc[0][s]);
      acc[1][s] = wmma_bf16(a1, b, acc[1][s]);
    }

    if (more) *(uint4*)(bdst + (cur ^ 1) * (64 * LDB)) = bnext;
    __syncthreads();
  }
}

// ---------------------------------------------------------------------------
// QKV GEMM: out[m,o] = sum_c x[m,c]*w[o,c] + b[o]
// Scatter:  q,k -> [b,h,n,d] bf16 ;  v -> [b,h,d,n] bf16 (pre-transposed).
// ---------------------------------------------------------------------------
__global__ __launch_bounds__(256)
void qkv_gemm_kernel(const unsigned short* __restrict__ xb,   // [8192,768] bf16
                     const unsigned short* __restrict__ wb,   // [2304,768] bf16
                     const float*          __restrict__ bias, // [2304] f32
                     unsigned short* __restrict__ qb,
                     unsigned short* __restrict__ kb,
                     unsigned short* __restrict__ vb)
{
  __shared__ unsigned short Bs[2 * 64 * LDB];          // 10 KB
  const int lane = threadIdx.x & 31;
  const int wave = threadIdx.x >> 5;
  const int h = lane >> 4, ln = lane & 15;

  const int bm    = blockIdx.x & 31;                    // 32 M-blocks of 256 rows
  const int tn    = blockIdx.x >> 5;                    // 0..35
  const int obase = tn * 64;
  const int row0  = bm * 256 + wave * 32;

  v8f acc[2][4] = {};
  gemm_core(xb, wb, row0, obase, Bs, acc);

  // ----- scatter with fused bias (which/head uniform per block) -----
  const int which = obase / DIM;                        // 0=q 1=k 2=v
  const int head  = (obase % DIM) / HD;

#pragma unroll
  for (int sub = 0; sub < 2; ++sub) {
    const int tok0 = row0 + sub * 16;
    const int bidx = tok0 / SEQ;                        // whole 16-row tile in one batch
    const int nq0  = (tok0 % SEQ) + 8 * h;
    if (which < 2) {
      unsigned short* dst = (which == 0) ? qb : kb;
#pragma unroll
      for (int s = 0; s < 4; ++s) {
        const int d = s * 16 + ln;
        const float bv = bias[obase + s * 16 + ln];
        size_t base = ((size_t)(bidx * NHEAD + head) * SEQ + nq0) * HD + d;
#pragma unroll
        for (int r = 0; r < 8; ++r)
          dst[base + (size_t)r * HD] = f2bf(acc[sub][s][r] + bv);
      }
    } else {                                            // V transposed: [b,h,d,n]
#pragma unroll
      for (int s = 0; s < 4; ++s) {
        const int d = s * 16 + ln;
        const float bv = bias[obase + s * 16 + ln];
        Pack8 pk;
#pragma unroll
        for (int r = 0; r < 8; ++r) pk.s[r] = f2bf(acc[sub][s][r] + bv);
        size_t base = ((size_t)(bidx * NHEAD + head) * HD + d) * SEQ + nq0;
        *(uint4*)(vb + base) = pk.u;                    // 8 consecutive tokens
      }
    }
  }
}

// ---------------------------------------------------------------------------
// Flash attention. One wave per 16-query tile per (b,h).
// S^T = K * Q^T  (softmax stats index by query = lane%16)
// O^T = V^T * P^T (accumulator rescale is a pure per-lane scalar multiply)
// ---------------------------------------------------------------------------
__global__ __launch_bounds__(256)
void attn_kernel(const unsigned short* __restrict__ qb,   // [bh,n,d]
                 const unsigned short* __restrict__ kb,   // [bh,n,d]
                 const unsigned short* __restrict__ vb,   // [bh,d,n]
                 unsigned short* __restrict__ sb)          // [b,n,768] bf16
{
  const int lane = threadIdx.x & 31;
  const int wave = threadIdx.x >> 5;
  const int gid  = blockIdx.x * 8 + wave;
  const int bh   = gid >> 6;                     // 0..95
  const int qt   = gid & 63;
  const int h    = lane >> 4;
  const int ln   = lane & 15;

  const unsigned short* Q = qb + (size_t)bh * SEQ * HD;
  const unsigned short* K = kb + (size_t)bh * SEQ * HD;
  const unsigned short* V = vb + (size_t)bh * HD * SEQ;

  const int q = qt * 16 + ln;                    // this lane's query column
  Frag bq0, bq1;                                 // Q as B operand (d = K-dim)
  bq0.u[0] = *(const uint4*)(Q + q * HD + 8 * h);
  bq0.u[1] = *(const uint4*)(Q + q * HD + 8 * h + 16);
  bq1.u[0] = *(const uint4*)(Q + q * HD + 32 + 8 * h);
  bq1.u[1] = *(const uint4*)(Q + q * HD + 32 + 8 * h + 16);

  v8f o0 = {}, o1 = {}, o2 = {}, o3 = {};        // O^T tiles, d-subtiles 0..3
  float m_run = -1e30f, l_run = 0.f;
  const float LOG2E = 1.44269504088896340736f;
  const float SCL   = 0.125f;                    // 64^-0.5

  for (int kt = 0; kt < SEQ; kt += 32) {
    // ---- S^T tiles: T0 = keys kt..kt+15, T1 = keys kt+16..kt+31 ----
    Frag a;
    const unsigned short* kr0 = K + (size_t)(kt + ln) * HD;
    const unsigned short* kr1 = K + (size_t)(kt + 16 + ln) * HD;
    v8f s0 = {}, s1 = {};
    a.u[0] = *(const uint4*)(kr0 + 8 * h);
    a.u[1] = *(const uint4*)(kr0 + 8 * h + 16);
    s0 = __builtin_amdgcn_wmma_f32_16x16x32_bf16(false, a.v, false, bq0.v, (short)0, s0, false, false);
    a.u[0] = *(const uint4*)(kr0 + 32 + 8 * h);
    a.u[1] = *(const uint4*)(kr0 + 32 + 8 * h + 16);
    s0 = __builtin_amdgcn_wmma_f32_16x16x32_bf16(false, a.v, false, bq1.v, (short)0, s0, false, false);
    a.u[0] = *(const uint4*)(kr1 + 8 * h);
    a.u[1] = *(const uint4*)(kr1 + 8 * h + 16);
    s1 = __builtin_amdgcn_wmma_f32_16x16x32_bf16(false, a.v, false, bq0.v, (short)0, s1, false, false);
    a.u[0] = *(const uint4*)(kr1 + 32 + 8 * h);
    a.u[1] = *(const uint4*)(kr1 + 32 + 8 * h + 16);
    s1 = __builtin_amdgcn_wmma_f32_16x16x32_bf16(false, a.v, false, bq1.v, (short)0, s1, false, false);

    // ---- online softmax (per query = ln; keys spread over vgprs + halves) ----
    float t0[8], t1[8], cmax = -1e30f;
#pragma unroll
    for (int r = 0; r < 8; ++r) {
      t0[r] = s0[r] * SCL;  t1[r] = s1[r] * SCL;
      cmax = fmaxf(cmax, fmaxf(t0[r], t1[r]));
    }
    cmax = fmaxf(cmax, __shfl_xor(cmax, 16, 32));
    const float m_new = fmaxf(m_run, cmax);
    const float alpha = exp2f((m_run - m_new) * LOG2E);
    float p0[8], p1[8], psum = 0.f;
#pragma unroll
    for (int r = 0; r < 8; ++r) {
      p0[r] = exp2f((t0[r] - m_new) * LOG2E);
      p1[r] = exp2f((t1[r] - m_new) * LOG2E);
      psum += p0[r] + p1[r];
    }
    psum += __shfl_xor(psum, 16, 32);
    l_run = l_run * alpha + psum;
    m_run = m_new;

    // ---- pack P^T as B operand (half exchange: one shfl_xor(16) per row) ----
    Frag pb;
#pragma unroll
    for (int r = 0; r < 8; ++r) {
      float send = h ? p0[r] : p1[r];
      float got  = __shfl_xor(send, 16, 32);
      pb.s[r]     = f2bf(h ? got : p0[r]);       // elems 0..7  -> keys 16h + r
      pb.s[8 + r] = f2bf(h ? p1[r] : got);       // elems 8..15 -> keys 16h + 8 + r
    }

    // ---- O^T += V^T * P^T  (4 d-subtiles) ----
#pragma unroll
    for (int t = 0; t < 4; ++t) {
      v8f& o = (t == 0) ? o0 : (t == 1) ? o1 : (t == 2) ? o2 : o3;
      o = o * alpha;
      Frag va;                                   // V^T rows contiguous over keys
      const unsigned short* vr = V + (size_t)(t * 16 + ln) * SEQ + kt;
      va.u[0] = *(const uint4*)(vr + 16 * h);
      va.u[1] = *(const uint4*)(vr + 16 * h + 8);
      o = __builtin_amdgcn_wmma_f32_16x16x32_bf16(false, va.v, false, pb.v, (short)0, o, false, false);
    }
  }

  // ---- normalize + store score (bf16, [b, n, h*64+d]) ----
  const float inv = 1.f / l_run;
  const int b  = bh / NHEAD;
  const int hh = bh % NHEAD;
  const size_t rowbase = ((size_t)(b * SEQ + q)) * DIM + hh * HD;
  v8f os[4] = {o0, o1, o2, o3};
#pragma unroll
  for (int t = 0; t < 4; ++t) {
    Pack8 pk;
#pragma unroll
    for (int r = 0; r < 8; ++r) pk.s[r] = f2bf(os[t][r] * inv);
    *(uint4*)(sb + rowbase + t * 16 + 8 * h) = pk.u; // 8 contiguous d -> 16B store
  }
}

// ---------------------------------------------------------------------------
// Output projection: out[m,o] = sum_c score[m,c]*w_proj[o,c] + b_proj[o], f32
// ---------------------------------------------------------------------------
__global__ __launch_bounds__(256)
void proj_gemm_kernel(const unsigned short* __restrict__ sb,   // [8192,768] bf16
                      const unsigned short* __restrict__ wb,   // [768,768]  bf16
                      const float*          __restrict__ bias, // [768] f32
                      float* __restrict__ out)                 // [8192,768] f32
{
  __shared__ unsigned short Bs[2 * 64 * LDB];          // 10 KB
  const int lane = threadIdx.x & 31;
  const int wave = threadIdx.x >> 5;
  const int h = lane >> 4, ln = lane & 15;

  const int bm    = blockIdx.x & 31;
  const int tn    = blockIdx.x >> 5;                    // 0..11
  const int obase = tn * 64;
  const int row0  = bm * 256 + wave * 32;

  v8f acc[2][4] = {};
  gemm_core(sb, wb, row0, obase, Bs, acc);

#pragma unroll
  for (int sub = 0; sub < 2; ++sub) {
#pragma unroll
    for (int s = 0; s < 4; ++s) {
      const int o = obase + s * 16 + ln;
      const float bv = bias[o];
      size_t base = (size_t)(row0 + sub * 16 + 8 * h) * DIM + o;
#pragma unroll
      for (int r = 0; r < 8; ++r)
        out[base + (size_t)r * DIM] = acc[sub][s][r] + bv;
    }
  }
}

// ---------------------------------------------------------------------------
// Host-side orchestration
// ---------------------------------------------------------------------------
extern "C" void kernel_launch(void* const* d_in, const int* in_sizes, int n_in,
                              void* d_out, int out_size, void* d_ws, size_t ws_size,
                              hipStream_t stream) {
  const float* x      = (const float*)d_in[0];   // [8,1024,768]
  const float* w_qkv  = (const float*)d_in[1];   // [2304,768]
  const float* b_qkv  = (const float*)d_in[2];   // [2304]
  const float* w_proj = (const float*)d_in[3];   // [768,768]
  const float* b_proj = (const float*)d_in[4];   // [768]
  float* out          = (float*)d_out;

  // workspace layout (bf16 buffers), ~64.5 MiB total
  char* ws = (char*)d_ws;
  unsigned short* xb     = (unsigned short*)(ws + 0);          // 8192*768
  unsigned short* wqkvb  = (unsigned short*)(ws + 12582912);   // 2304*768
  unsigned short* wprojb = (unsigned short*)(ws + 16121856);   // 768*768
  unsigned short* qb     = (unsigned short*)(ws + 17301504);   // [96,1024,64]
  unsigned short* kb     = (unsigned short*)(ws + 29884416);   // [96,1024,64]
  unsigned short* vb     = (unsigned short*)(ws + 42467328);   // [96,64,1024]
  unsigned short* sb     = (unsigned short*)(ws + 55050240);   // 8192*768

  const int nx4 = MTOK * DIM / 4;     // 1572864
  const int nw4 = QKVO * DIM / 4;     // 442368
  const int np4 = DIM * DIM / 4;      // 147456
  cvt_bf16_kernel<<<(nx4 + 255) / 256, 256, 0, stream>>>((const float4*)x, (ushort4*)xb, nx4);
  cvt_bf16_kernel<<<(nw4 + 255) / 256, 256, 0, stream>>>((const float4*)w_qkv, (ushort4*)wqkvb, nw4);
  cvt_bf16_kernel<<<(np4 + 255) / 256, 256, 0, stream>>>((const float4*)w_proj, (ushort4*)wprojb, np4);

  // QKV GEMM: 32 M-blocks x 36 N-blocks
  qkv_gemm_kernel<<<32 * 36, 256, 0, stream>>>(xb, wqkvb, b_qkv, qb, kb, vb);

  // Attention: 96 (b,h) * 64 q-tiles = 6144 wave-tiles -> 768 blocks
  attn_kernel<<<768, 256, 0, stream>>>(qb, kb, vb, sb);

  // Output projection: 32 M-blocks x 12 N-blocks
  proj_gemm_kernel<<<32 * 12, 256, 0, stream>>>(sb, wprojb, b_proj, out);
}
